// _AGCNRN__56478819942833
// MI455X (gfx1250) — matrix operand order repaired and checked by reading, not compiled, not run on hardware
//
#include <hip/hip_runtime.h>

// Problem constants (from reference setup_inputs)
#define N_NODES 5570
#define BATCH   16
#define C_IN    35
#define BC      560     // BATCH * C_IN (GEMM N-dim), 560 = 7*80
#define KPAD    5600    // k-dim padded to multiple of 32
#define RPAD    5632    // output rows padded to multiple of BM
#define BM      128     // GEMM row tile (8 waves x 16 rows)
#define BN      80      // GEMM col tile (5 WMMA col tiles)
#define KB      32      // k step (bf16 WMMA K)
#define NITER   (KPAD / KB)   // 175
#define LBS     40      // LDS B column stride in bf16 (32 data + 8 pad, bank-conflict-free)

typedef __attribute__((ext_vector_type(16))) __bf16   v16bf;
typedef __attribute__((ext_vector_type(8)))  __bf16   v8bf;
typedef __attribute__((ext_vector_type(8)))  float    v8f;
typedef __attribute__((ext_vector_type(4)))  unsigned v4u;
typedef __attribute__((ext_vector_type(4)))  unsigned u32x4;
typedef __attribute__((ext_vector_type(4)))  int      i32x4;
typedef __attribute__((ext_vector_type(8)))  int      i32x8;

// Use the Tensor Data Mover when the toolchain exposes it (device pass only);
// otherwise (and on the host pass) fall back to register-staged LDS tiles.
#if defined(__HIP_DEVICE_COMPILE__) && __has_builtin(__builtin_amdgcn_tensor_load_to_lds)
#define USE_TDM 1
#else
#define USE_TDM 0
#endif

__device__ __forceinline__ __bf16 f2bf(float f) {
  unsigned u = __builtin_bit_cast(unsigned, f);
  u += 0x7FFFu + ((u >> 16) & 1u);            // round-to-nearest-even
  unsigned short h = (unsigned short)(u >> 16);
  return __builtin_bit_cast(__bf16, h);
}

// ---------------------------------------------------------------------------
// Kernel 1: pack x [B,N,C] -> X^T bf16 [BC x KPAD]: Xbf[(b*C+c)*KPAD + m] = x[b,m,c]
// ---------------------------------------------------------------------------
__global__ void k_pack_x_bf16(const float* __restrict__ x, __bf16* __restrict__ Xbf) {
  int idx = blockIdx.x * blockDim.x + threadIdx.x;
  const int total = BC * KPAD;
  if (idx >= total) return;
  int col = idx / KPAD;
  int m   = idx - col * KPAD;
  int b   = col / C_IN;
  int c   = col - b * C_IN;
  float v = 0.f;
  if (m < N_NODES) v = x[((size_t)b * N_NODES + m) * C_IN + c];
  Xbf[idx] = f2bf(v);
}

// ---------------------------------------------------------------------------
// Kernel 2: per-row softmax(relu(E E^T)); store UNNORMALIZED exp as bf16,
// store 1/rowsum in fp32 (applied in GEMM epilogue). Padded rows/cols -> 0.
// ---------------------------------------------------------------------------
__global__ __launch_bounds__(256) void k_softmax(const float* __restrict__ e,
                                                 __bf16* __restrict__ Sexp,
                                                 float* __restrict__ rowscale) {
  __shared__ float red[256];
  const int n   = blockIdx.x;
  const int tid = threadIdx.x;
  __bf16* srow = Sexp + (size_t)n * KPAD;
  if (n >= N_NODES) {                       // zero-pad rows (block-uniform branch)
    for (int m = tid; m < KPAD; m += 256) srow[m] = f2bf(0.f);
    if (tid == 0) rowscale[n] = 0.f;
    return;
  }
  const float4* e4 = (const float4*)e;      // E is [N,4], 16B rows
  float4 en = e4[n];
  float maxv = 0.f;                         // relu -> logits >= 0
  for (int m = tid; m < N_NODES; m += 256) {
    float4 em = e4[m];
    float d = fmaxf(en.x*em.x + en.y*em.y + en.z*em.z + en.w*em.w, 0.f);
    maxv = fmaxf(maxv, d);
  }
  red[tid] = maxv;
  __syncthreads();
  for (int s = 128; s > 0; s >>= 1) {
    if (tid < s) red[tid] = fmaxf(red[tid], red[tid + s]);
    __syncthreads();
  }
  maxv = red[0];
  __syncthreads();
  float sum = 0.f;
  for (int m = tid; m < KPAD; m += 256) {
    float ev = 0.f;
    if (m < N_NODES) {
      float4 em = e4[m];
      float d = fmaxf(en.x*em.x + en.y*em.y + en.z*em.z + en.w*em.w, 0.f);
      ev = __expf(d - maxv);
    }
    srow[m] = f2bf(ev);
    sum += ev;
  }
  red[tid] = sum;
  __syncthreads();
  for (int s = 128; s > 0; s >>= 1) {
    if (tid < s) red[tid] += red[tid + s];
    __syncthreads();
  }
  if (tid == 0) rowscale[n] = 1.f / red[0];
}

// ---------------------------------------------------------------------------
// TDM: one wave DMAs a 32(k) x 80(col) bf16 tile of X^T into LDS.
// Hardware padding: +16B after every 64B row  =>  LDS column stride LBS=40.
// ---------------------------------------------------------------------------
#if USE_TDM
__device__ __forceinline__ void tdm_load_B(const __bf16* gtile, unsigned ldsOff) {
  unsigned long long ga = (unsigned long long)(uintptr_t)gtile;
  u32x4 g0;
  g0[0] = 1u;                                                // count=1, user mode
  g0[1] = ldsOff;                                            // lds_addr (bytes)
  g0[2] = (unsigned)ga;                                      // global_addr[31:0]
  g0[3] = ((unsigned)(ga >> 32) & 0x1FFFFFFu) | (2u << 30);  // global_addr[56:32], type=2
  i32x8 g1;
  g1[0] = (int)((1u << 16)      // data_size = 2 bytes
              | (1u << 20)      // pad_enable
              | (3u << 22)      // pad_interval: 16 DWORDs = 64B
              | (3u << 25));    // pad_amount: 4 DWORDs = 16B
  g1[1] = (int)(5600u << 16);   // tensor_dim0[15:0] = KPAD
  g1[2] = (int)(560u << 16);    // tensor_dim1[15:0] = BC
  g1[3] = (int)(32u << 16);     // tile_dim0 = KB
  g1[4] = (int)80u;             // tile_dim1 = BN, tile_dim2 = 0
  g1[5] = (int)5600u;           // tensor_dim0_stride = KPAD
  g1[6] = 0;
  g1[7] = 0;
  i32x4 z4 = {0, 0, 0, 0};      // 2-D tile: groups 2/3 unused
#if __clang_major__ >= 23
  i32x8 z8 = {0, 0, 0, 0, 0, 0, 0, 0};
  __builtin_amdgcn_tensor_load_to_lds(g0, g1, z4, z4, z8, 0);
#else
  __builtin_amdgcn_tensor_load_to_lds(g0, g1, z4, z4, 0);
#endif
}
#endif

// ---------------------------------------------------------------------------
// Kernel 3: P = (Sexp @ X) * rowscale via v_wmma_f32_16x16x32_bf16.
// Block = 256 threads (8 waves), tile 128x80; wave w -> rows [16w,16w+16).
// A fragments loaded DIRECTLY from global (two b128/lane; S is L2-resident).
// B tile in LDS via TDM (tensor_load_to_lds) or reg staging; double-buffered,
// software-pipelined, one barrier per k-step.
// ---------------------------------------------------------------------------
union AFrag { v16bf v; v4u q[2]; };

__global__ __launch_bounds__(256) void k_gemm_wmma(
    const __bf16* __restrict__ S, const __bf16* __restrict__ X,
    const float* __restrict__ rowscale, float* __restrict__ P) {
  __shared__ __bf16 lB[2][BN * LBS];   // [col][k] with padded stride
  const int tid  = threadIdx.x;
  const int lane = tid & 31;
  const int wave = tid >> 5;
  const int half = lane >> 4;
  const int l16  = lane & 15;
  const int rowBase = blockIdx.x * BM;
  const int colBase = blockIdx.y * BN;

  // Per-lane A row pointer (16-bit A 16x32 layout: lane M=l16, half picks K-groups)
  const __bf16* pA = S + (size_t)(rowBase + (wave << 4) + l16) * KPAD;
  const __bf16* Xcol = X + (size_t)colBase * KPAD;   // block's B panel

  v8f acc[5];
  #pragma unroll
  for (int t = 0; t < 5; ++t)
    #pragma unroll
    for (int j = 0; j < 8; ++j) acc[t][j] = 0.f;

#if !USE_TDM
  // B-tile chunk assignment: 80 cols x 64B = 320 x 16B chunks, 256 threads
  const int c0_col  = tid >> 2;
  const int c0_kOff = (tid & 3) << 3;
  const int c1_col  = (tid + 256) >> 2;        // only threads < 64 use this
  const int c1_kOff = (tid & 3) << 3;
  const __bf16* pB0 = Xcol + (size_t)c0_col * KPAD + c0_kOff;
  const __bf16* pB1 = Xcol + (size_t)c1_col * KPAD + c1_kOff;
#else
  const unsigned ldsB[2] = { (unsigned)(uintptr_t)(void*)&lB[0][0],
                             (unsigned)(uintptr_t)(void*)&lB[1][0] };
#endif

  // ---- prologue: A frag 0 + B tile 0 ----
  AFrag ua;
  ua.q[0] = *(const v4u*)(pA + (half << 3));
  ua.q[1] = *(const v4u*)(pA + 16 + (half << 3));
#if USE_TDM
  if (wave == 0) {
    tdm_load_B(Xcol, ldsB[0]);
    __builtin_amdgcn_s_wait_tensorcnt(0);
  }
#else
  {
    v8bf b0 = *(const v8bf*)pB0;
    *(v8bf*)(&lB[0][c0_col * LBS + c0_kOff]) = b0;
    if (tid < 64) {
      v8bf b1 = *(const v8bf*)pB1;
      *(v8bf*)(&lB[0][c1_col * LBS + c1_kOff]) = b1;
    }
  }
#endif
  __syncthreads();

  for (int it = 0; it < NITER; ++it) {
    const int kb  = it * KB;
    const int cur = it & 1;
    const int nxt = cur ^ 1;
    const bool hasNext = (it + 1) < NITER;
    const int kn = kb + KB;

    // issue next-tile loads early (overlap with WMMA)
    AFrag uan;
#if !USE_TDM
    v8bf b0, b1;
#endif
    if (hasNext) {
#if USE_TDM
      if (wave == 0) tdm_load_B(Xcol + kn, ldsB[nxt]);
#else
      b0 = *(const v8bf*)(pB0 + kn);
      if (tid < 64) b1 = *(const v8bf*)(pB1 + kn);
#endif
      uan.q[0] = *(const v4u*)(pA + kn + (half << 3));
      uan.q[1] = *(const v4u*)(pA + kn + 16 + (half << 3));
      // speculative prefetch two steps ahead (global_prefetch_b8)
      if (kb + 2 * KB < KPAD) __builtin_prefetch(pA + kb + 2 * KB, 0, 0);
    }

    // compute: 5 column tiles against current B buffer
    #pragma unroll
    for (int ct = 0; ct < 5; ++ct) {
      // B frag (32x16): lane N=l16; K at half*8 and 16+half*8 (8 each)
      union { v16bf v; v4u q[2]; } ub;
      const __bf16* cp = &lB[cur][((ct << 4) + l16) * LBS + (half << 3)];
      ub.q[0] = *(const v4u*)cp;
      ub.q[1] = *(const v4u*)(cp + 16);
      acc[ct] = __builtin_amdgcn_wmma_f32_16x16x32_bf16(
          false, ua.v, false, ub.v, (short)0, acc[ct], false, false);
    }

    // commit next B tile, then one barrier
    if (hasNext) {
#if USE_TDM
      if (wave == 0) __builtin_amdgcn_s_wait_tensorcnt(0);
#else
      *(v8bf*)(&lB[nxt][c0_col * LBS + c0_kOff]) = b0;
      if (tid < 64) *(v8bf*)(&lB[nxt][c1_col * LBS + c1_kOff]) = b1;
#endif
    }
    __syncthreads();
    ua = uan;
  }

  // Epilogue: C/D layout -> element (M = r + 8*half, N = l16); apply 1/rowsum.
  float rs[8];
  #pragma unroll
  for (int r = 0; r < 8; ++r)
    rs[r] = rowscale[rowBase + (wave << 4) + (half << 3) + r];
  #pragma unroll
  for (int ct = 0; ct < 5; ++ct)
    #pragma unroll
    for (int r = 0; r < 8; ++r) {
      int row = rowBase + (wave << 4) + (half << 3) + r;
      int col = colBase + (ct << 4) + l16;
      P[(size_t)row * BC + col] = acc[ct][r] * rs[r];
    }
}

// ---------------------------------------------------------------------------
// Kernel 4: fused GRU epilogue. H=0 => only R (gate o=2,3) and C matter:
//   h = (1-R)*tanh(update);  y = relu(h) @ lin_w + lin_b
// ---------------------------------------------------------------------------
__global__ void k_final(const float* __restrict__ x, const float* __restrict__ e,
                        const float* __restrict__ w_gate, const float* __restrict__ b_gate,
                        const float* __restrict__ w_update, const float* __restrict__ b_update,
                        const float* __restrict__ lin_w, const float* __restrict__ lin_b,
                        const float* __restrict__ P, float* __restrict__ out) {
  int idx = blockIdx.x * blockDim.x + threadIdx.x;
  if (idx >= BATCH * N_NODES) return;
  int b = idx / N_NODES;
  int n = idx - b * N_NODES;
  float e0 = e[n*4+0], e1 = e[n*4+1], e2 = e[n*4+2], e3 = e[n*4+3];
  float r0 = e0*b_gate[0*4+2] + e1*b_gate[1*4+2] + e2*b_gate[2*4+2] + e3*b_gate[3*4+2];
  float r1 = e0*b_gate[0*4+3] + e1*b_gate[1*4+3] + e2*b_gate[2*4+3] + e3*b_gate[3*4+3];
  float u0 = e0*b_update[0*2+0] + e1*b_update[1*2+0] + e2*b_update[2*2+0] + e3*b_update[3*2+0];
  float u1 = e0*b_update[0*2+1] + e1*b_update[1*2+1] + e2*b_update[2*2+1] + e3*b_update[3*2+1];
  const float* xr = x + ((size_t)b * N_NODES + n) * C_IN;
  const float* pr = P + (size_t)n * BC + b * C_IN;
  #define WGD(d,k,o) w_gate[(((d)*2+(k))*(C_IN+2) + i)*4 + (o)]
  #define WUD(d,k,o) w_update[(((d)*2+(k))*(C_IN+2) + i)*2 + (o)]
  for (int i = 0; i < C_IN; ++i) {
    float xv = xr[i];
    float pv = pr[i];
    float g02 = e0*WGD(0,0,2)+e1*WGD(1,0,2)+e2*WGD(2,0,2)+e3*WGD(3,0,2);
    float g12 = e0*WGD(0,1,2)+e1*WGD(1,1,2)+e2*WGD(2,1,2)+e3*WGD(3,1,2);
    float g03 = e0*WGD(0,0,3)+e1*WGD(1,0,3)+e2*WGD(2,0,3)+e3*WGD(3,0,3);
    float g13 = e0*WGD(0,1,3)+e1*WGD(1,1,3)+e2*WGD(2,1,3)+e3*WGD(3,1,3);
    r0 += xv*g02 + pv*g12;
    r1 += xv*g03 + pv*g13;
    float q00 = e0*WUD(0,0,0)+e1*WUD(1,0,0)+e2*WUD(2,0,0)+e3*WUD(3,0,0);
    float q10 = e0*WUD(0,1,0)+e1*WUD(1,1,0)+e2*WUD(2,1,0)+e3*WUD(3,1,0);
    float q01 = e0*WUD(0,0,1)+e1*WUD(1,0,1)+e2*WUD(2,0,1)+e3*WUD(3,0,1);
    float q11 = e0*WUD(0,1,1)+e1*WUD(1,1,1)+e2*WUD(2,1,1)+e3*WUD(3,1,1);
    u0 += xv*q00 + pv*q10;
    u1 += xv*q01 + pv*q11;
  }
  #undef WGD
  #undef WUD
  float R0 = 1.f / (1.f + __expf(-r0));
  float R1 = 1.f / (1.f + __expf(-r1));
  float h0 = (1.f - R0) * tanhf(u0);
  float h1 = (1.f - R1) * tanhf(u1);
  float y = fmaxf(h0, 0.f) * lin_w[0] + fmaxf(h1, 0.f) * lin_w[1] + lin_b[0];
  out[idx] = y;   // [B,N,1] flat = b*N + n
}

// ---------------------------------------------------------------------------
extern "C" void kernel_launch(void* const* d_in, const int* in_sizes, int n_in,
                              void* d_out, int out_size, void* d_ws, size_t ws_size,
                              hipStream_t stream) {
  const float* x        = (const float*)d_in[0];
  const float* e        = (const float*)d_in[1];
  const float* w_gate   = (const float*)d_in[2];
  const float* b_gate   = (const float*)d_in[3];
  const float* w_update = (const float*)d_in[4];
  const float* b_update = (const float*)d_in[5];
  const float* lin_w    = (const float*)d_in[6];
  const float* lin_b    = (const float*)d_in[7];
  float* out = (float*)d_out;
  (void)in_sizes; (void)n_in; (void)out_size; (void)ws_size;

  // Workspace layout (~82 MB total)
  char* ws = (char*)d_ws;
  size_t off = 0;
  __bf16* Sexp    = (__bf16*)(ws + off); off += (size_t)RPAD * KPAD * 2;  // 63.1 MB
  float* rowscale = (float*) (ws + off); off += (size_t)RPAD * 4;
  __bf16* Xbf     = (__bf16*)(ws + off); off += (size_t)BC * KPAD * 2;    // 6.3 MB (X^T)
  float* P        = (float*) (ws + off); off += (size_t)RPAD * BC * 4;    // 12.6 MB

  {
    int total = BC * KPAD;
    k_pack_x_bf16<<<(total + 255) / 256, 256, 0, stream>>>(x, Xbf);
  }
  k_softmax<<<RPAD, 256, 0, stream>>>(e, Sexp, rowscale);
  {
    dim3 g(RPAD / BM, BC / BN);   // 44 x 7
    k_gemm_wmma<<<g, 256, 0, stream>>>(Sexp, Xbf, rowscale, P);
  }
  {
    int total = BATCH * N_NODES;
    k_final<<<(total + 255) / 256, 256, 0, stream>>>(x, e, w_gate, b_gate,
                                                     w_update, b_update,
                                                     lin_w, lin_b, P, out);
  }
}
